// FFTResonanceBlock_89034672046353
// MI455X (gfx1250) — matrix operand order repaired and checked by reading, not compiled, not run on hardware
//
#include <hip/hip_runtime.h>
#include <hip/hip_bf16.h>
#include <stdint.h>

typedef __attribute__((ext_vector_type(16))) _Float16 v16h;
typedef __attribute__((ext_vector_type(8)))  float    v8f;

#define N_SAMPLES 65536
#define WINDOW    1024
#define STEP      512
#define N_COEFFS  513
#define N_FRAMES  128
#define N_RES     64
#define EXPR      4
#define KPAD      544            // 513 padded to 17*32 (per re/im half)
#define KTOT      1088           // 2 * KPAD
#define NCHUNK    34             // KTOT / 32
#define NRE       (N_RES * EXPR) // 256

static constexpr float PI_F = 3.14159265358979323846f;

// ---------------------------------------------------------------- zero output
__global__ void zero_out_kernel(float4* __restrict__ out, int n4) {
    int i = blockIdx.x * blockDim.x + threadIdx.x;
    if (i < n4) out[i] = make_float4(0.f, 0.f, 0.f, 0.f);
}

// ------------------------------------------------- per (r,e,k) parameter pack
// P[(r*4+e)*KPAD + kk] = { log(|amp|+eps), log(decay_mag), cos(phi), sin(phi) }
__global__ void build_params_kernel(const float* __restrict__ amp,
                                    const float* __restrict__ phase,
                                    const float* __restrict__ decay,
                                    float4* __restrict__ P) {
    int idx = blockIdx.x * blockDim.x + threadIdx.x;
    if (idx >= NRE * KPAD) return;
    int re = idx / KPAD;
    int kk = idx - re * KPAD;
    float4 o = make_float4(0.f, 0.f, 0.f, 0.f);   // pad rows -> A = exp(0)*0 = 0
    if (kk < N_COEFFS) {
        int r = re >> 2, e = re & 3;
        int src = (r * N_COEFFS + kk) * EXPR + e;   // input layout (R, K, E)
        float a  = fabsf(amp[src]);
        float d  = decay[src];
        float ph = phase[src];
        float sig = 1.0f / (1.0f + __expf(-d));
        float dm  = 0.5f + sig * (0.5f * 0.9f);     // BASE_RES + sigmoid*span*RES_FACTOR
        float sp  = tanhf(ph) * PI_F;
        o.x = __logf(a + 1e-12f);
        o.y = __logf(dm);
        o.z = cosf(sp);
        o.w = sinf(sp);
    }
    P[idx] = o;
}

// -------------------------- irfft+window basis, stored in WMMA-B frag order
// B_re[k][n] = (w_k/1024) cos(2pi k n / 1024) * win[n]
// B_im[k][n] = -(w_k/1024) sin(2pi k n / 1024) * win[n]
// swizzle: u32 slot = ((((c*64 + n>>4)*2 + khalf)*16 + (n&15))*8 + j), half = k&1
__global__ void build_basis_kernel(_Float16* __restrict__ Bh) {
    int idx = blockIdx.x * blockDim.x + threadIdx.x;
    if (idx >= KTOT * WINDOW) return;
    int k = idx >> 10;
    int n = idx & (WINDOW - 1);
    int isIm = (k >= KPAD);
    int kk = isIm ? (k - KPAD) : k;
    float val = 0.f;
    if (kk < N_COEFFS) {
        float w = (kk == 0 || kk == 512) ? 1.0f : 2.0f;
        int p = (kk * n) & (WINDOW - 1);                     // exact mod-2pi reduction
        float ang = (2.0f * PI_F / (float)WINDOW) * (float)p;
        float win = 0.5f - 0.5f * cosf((2.0f * PI_F / (float)WINDOW) * (float)n);
        float tr  = isIm ? -sinf(ang) : cosf(ang);
        val = tr * w * (1.0f / (float)WINDOW) * win;
    }
    int c  = k >> 5;
    int ki = k & 31;
    int kh = ki >> 4;
    int j  = (ki & 15) >> 1;
    int t  = ki & 1;
    int idx32 = ((((c * 64) + (n >> 4)) * 2 + kh) * 16 + (n & 15)) * 8 + j;
    Bh[idx32 * 2 + t] = (_Float16)val;
}

// -------------------------------------------------------------- WMMA GEMM
// grid = NRE * 8 workgroups; each WG: fixed (r,e), 16-frame tile, full 1024 cols.
// 8 waves/WG, each wave owns 128 columns (8 WMMA N-tiles).
__global__ __launch_bounds__(256) void synth_gemm_kernel(
        const float4*   __restrict__ params,
        const uint32_t* __restrict__ Bbuf,
        float*          __restrict__ out) {
    __shared__ float4 P[KPAD];

    const int wg = blockIdx.x;
    const int re = wg >> 3;               // 0..255 -> (r*4 + e)
    const int f0 = (wg & 7) << 4;         // frame-tile base

    const int tid = threadIdx.x;
    for (int i = tid; i < KPAD; i += 256) P[i] = params[re * KPAD + i];
    __syncthreads();

    const int lane  = tid & 31;
    const int wid   = tid >> 5;           // wave id -> columns [wid*128, wid*128+128)
    const int khalf = lane >> 4;
    const int row   = lane & 15;          // A row within tile / B column within tile
    const int fA    = f0 + row;           // frame index of this lane's A row
    const float ff  = (float)fA;

    v8f acc[8] = {};

    for (int c = 0; c < NCHUNK; ++c) {
        // ---- build A fragment (16x32 f16) in registers
        v16h a;
#pragma unroll
        for (int j = 0; j < 8; ++j) {
#pragma unroll
            for (int t = 0; t < 2; ++t) {
                int kin = (khalf << 4) + (j << 1) + t;
                int k   = (c << 5) + kin;
                int isIm = (k >= KPAD);
                int kk   = isIm ? (k - KPAD) : k;
                float4 p = P[kk];                     // ds_load_b128
                float mag = __expf(p.x + ff * p.y);   // v_exp_f32
                float tr  = isIm ? p.w : p.z;
                float v   = mag * tr;
                v = (fA & kk & 1) ? -v : v;           // (-1)^(f*k)
                a[2 * j + t] = (_Float16)v;
            }
        }
        // ---- 8 column tiles: load B frag (2x global_load_b128) + WMMA
#pragma unroll
        for (int nt = 0; nt < 8; ++nt) {
            int ngrp = (wid << 3) + nt;               // n >> 4
            const v16h* bp = (const v16h*)(Bbuf +
                (((((c * 64) + ngrp) * 2 + khalf) * 16 + row) << 3));
            v16h b = *bp;
            acc[nt] = __builtin_amdgcn_wmma_f32_16x16x32_f16(
                false, a, false, b, (short)0, acc[nt], false, false);
        }
    }

    // ---- overlap-add write-out: each frame element hits exactly one atomic
    float* base = out + re * N_SAMPLES;
#pragma unroll
    for (int nt = 0; nt < 8; ++nt) {
        int n = (wid << 7) + (nt << 4) + row;         // column within frame (0..1023)
#pragma unroll
        for (int r = 0; r < 8; ++r) {
            int fr = f0 + r + (khalf << 3);           // D-matrix row -> frame
            float v = acc[nt][r];
            if (n < STEP) {
                atomicAdd(base + fr * STEP + n, v);
            } else if (fr < N_FRAMES - 1) {           // last frame's 2nd half dropped
                atomicAdd(base + (fr + 1) * STEP + (n - STEP), v);
            }
        }
    }
}

// ---------------------------------------------------------------- launcher
extern "C" void kernel_launch(void* const* d_in, const int* in_sizes, int n_in,
                              void* d_out, int out_size, void* d_ws, size_t ws_size,
                              hipStream_t stream) {
    const float* amp   = (const float*)d_in[0];
    const float* phase = (const float*)d_in[1];
    const float* decay = (const float*)d_in[2];
    float* out = (float*)d_out;

    // workspace layout
    const size_t B_BYTES = (size_t)NCHUNK * 64 * 2 * 16 * 8 * 4;   // 2,228,224
    uint32_t* Bbuf = (uint32_t*)d_ws;
    float4*   Pbuf = (float4*)((char*)d_ws + B_BYTES);             // 2,228,224 more

    // 1) zero output (harness poisons it; atomics need 0 base)
    {
        int n4 = out_size / 4;
        int blocks = (n4 + 255) / 256;
        zero_out_kernel<<<blocks, 256, 0, stream>>>((float4*)out, n4);
    }
    // 2) per-(r,e,k) parameters
    {
        int total = NRE * KPAD;
        build_params_kernel<<<(total + 255) / 256, 256, 0, stream>>>(amp, phase, decay, Pbuf);
    }
    // 3) swizzled f16 irfft+window basis
    {
        int total = KTOT * WINDOW;
        build_basis_kernel<<<(total + 255) / 256, 256, 0, stream>>>((_Float16*)Bbuf);
    }
    // 4) WMMA GEMM + overlap-add
    synth_gemm_kernel<<<NRE * 8, 256, 0, stream>>>(Pbuf, Bbuf, out);
}
